// RNN_4286377361416
// MI455X (gfx1250) — matrix operand order repaired and checked by reading, not compiled, not run on hardware
//
#include <hip/hip_runtime.h>

// ---------------------------------------------------------------------------
// GRU scan on MI455X (gfx1250).
// Latency-bound serial recurrence: per step gh = h @ Whh^T is a [32,64]x[64,192]
// GEMM = 24 WMMA 16x16 tiles with K=64 (2 chained v_wmma_f32_16x16x32_f16).
// 2 workgroups (batch tiles of 16) x 12 waves (one gate tile each).
// Whh held in registers as f16 B-fragments; h carried in LDS (f16, A-frags)
// and in f32 registers (epilogue waves). x staged via async global->LDS.
// In-loop barriers are LDS-only (DScnt) so streamed global `states` stores
// never stall the recurrence chain on STOREcnt drains.
// ---------------------------------------------------------------------------

constexpr int Bb = 32, Tt = 16384, Hh = 64;
#define WAVES 12
#define TB (WAVES * 32)

typedef __attribute__((ext_vector_type(16))) _Float16 v16h;
typedef __attribute__((ext_vector_type(8)))  _Float16 h8;
typedef __attribute__((ext_vector_type(8)))  float    v8f;

union V16 { v16h v; h8 h[2]; };

__device__ __forceinline__ float fast_exp(float x) {
#if __has_builtin(__builtin_amdgcn_exp2f)
  return __builtin_amdgcn_exp2f(x * 1.4426950408889634f);
#else
  return __expf(x);
#endif
}
__device__ __forceinline__ float fast_rcp(float x) {
#if __has_builtin(__builtin_amdgcn_rcpf)
  return __builtin_amdgcn_rcpf(x);
#else
  return 1.0f / x;
#endif
}
__device__ __forceinline__ float sigmoid_(float a) {
  return fast_rcp(1.0f + fast_exp(-a));
}
__device__ __forceinline__ float tanh_(float a) {
#if __has_builtin(__builtin_amdgcn_tanhf)
  return __builtin_amdgcn_tanhf(a);            // gfx1250 v_tanh_f32
#else
  return 1.0f - 2.0f * fast_rcp(fast_exp(2.0f * a) + 1.0f);
#endif
}

// LDS-only workgroup barrier: waits DScnt (LDS visibility across the WGP) but
// deliberately does NOT drain STOREcnt, so fire-and-forget global stores of
// `states` stay off the serial critical path. __syncthreads() would emit
// s_wait_storecnt_dscnt 0x0 here (observed in disassembly), costing a GL2
// store-ack round trip every timestep.
__device__ __forceinline__ void lds_barrier() {
  asm volatile("s_wait_dscnt 0x0\n\t"
               "s_barrier_signal -1\n\t"
               "s_barrier_wait -1" ::: "memory");
}

#if __has_builtin(__builtin_amdgcn_global_load_async_to_lds_b32) && \
    __has_builtin(__builtin_amdgcn_s_wait_asynccnt)
#define HAVE_ASYNC_LDS 1
typedef __attribute__((address_space(1))) int gas_int;   // global (AS1)
typedef __attribute__((address_space(3))) int las_int;   // LDS (AS3)
__device__ __forceinline__ void async_g2l_b32(const float* g, float* l) {
  __builtin_amdgcn_global_load_async_to_lds_b32((gas_int*)g, (las_int*)l, 0, 0);
}
#else
#define HAVE_ASYNC_LDS 0
#endif

__global__ void __launch_bounds__(TB, 1) gru_scan_kernel(
    const float* __restrict__ x,   const float* __restrict__ Wih,
    const float* __restrict__ Whh, const float* __restrict__ bih,
    const float* __restrict__ bhh, float* __restrict__ states)
{
  __shared__ _Float16 lds_h16[16][64] __attribute__((aligned(16))); // h state (f16)
  __shared__ float    lds_gh[16][128];                              // z,n gh tiles
  __shared__ float    lds_x[2][64][16];                             // staged x

  const int tid  = threadIdx.x;
  const int w    = tid >> 5;      // wave id 0..11 -> gate-col tile
  const int lane = tid & 31;
  const int lr   = lane & 15;     // row (A/M) or col (B/N, D/N) within tile
  const int lh   = lane >> 4;     // half-wave selector
  const int b0   = blockIdx.x * 16;

  // ---- loop-invariant B fragments: Whh^T tile for this wave's 16 gate cols.
  // B 32x16 f16 frag (code's K order, same as A): lanes 0-15 hold K=0..15, lanes 16-31 K=16..31
  // (two halves per VGPR) -> 16 contiguous K per lane = Whh row-major slice.
  const int g = w * 16 + lr;      // gate output index in [0,192)
  V16 bf_lo, bf_hi;
  {
    const float* wrow = Whh + g * Hh + lh * 16;
#pragma unroll
    for (int i = 0; i < 16; ++i) {
      bf_lo.v[i] = (_Float16)wrow[i];        // K = lh*16 + i
      bf_hi.v[i] = (_Float16)wrow[32 + i];   // K = 32 + lh*16 + i
    }
  }
  const float bias_hh = bhh[g];

  // ---- epilogue constants: waves 0..3 own hidden column c = g in [0,64)
  float wir = 0, wiz = 0, win = 0, bir = 0, biz = 0, bin_ = 0;
  float hreg[8] = {0, 0, 0, 0, 0, 0, 0, 0};   // f32 h carried in registers
  const int c = g;
  if (w < 4) {
    wir = Wih[c];        bir  = bih[c];
    wiz = Wih[64 + c];   biz  = bih[64 + c];
    win = Wih[128 + c];  bin_ = bih[128 + c];
  }

  // zero h0 in LDS
  for (int i = tid; i < 16 * 64; i += TB) ((_Float16*)lds_h16)[i] = (_Float16)0.0f;
  __syncthreads();

  const int rbase = lh * 8;       // D-matrix rows: lanes 0-15 -> M=j, 16-31 -> M=8+j

#if HAVE_ASYNC_LDS
  // prime: async-stage x tile 0 into buffer 0 (waves 8..11, ASYNCcnt tracked)
  if (w >= 8) {
    int q = tid - 256;
#pragma unroll
    for (int i = 0; i < 8; ++i) {
      int e = q * 8 + i, row = e & 15, tt = e >> 4;
      async_g2l_b32(x + (b0 + row) * Tt + tt, &lds_x[0][tt][row]);
    }
  }
#endif

  for (int t = 0; t < Tt; ++t) {
    // ---- phase A: gh = h @ Whh^T + bhh, one 16x16 tile per wave ----------
    // A 16x32 f16 frag (code's K order, same as B): lane m=lr, 16 contiguous K per half-wave (K = lh*16 + s).
    V16 a_lo, a_hi;
    a_lo.h[0] = *(const h8*)&lds_h16[lr][lh * 16];
    a_lo.h[1] = *(const h8*)&lds_h16[lr][lh * 16 + 8];
    a_hi.h[0] = *(const h8*)&lds_h16[lr][32 + lh * 16];
    a_hi.h[1] = *(const h8*)&lds_h16[lr][40 + lh * 16];

    v8f acc = {bias_hh, bias_hh, bias_hh, bias_hh,
               bias_hh, bias_hh, bias_hh, bias_hh};
    acc = __builtin_amdgcn_wmma_f32_16x16x32_f16(false, a_lo.v, false, bf_lo.v,
                                                 (short)0, acc, false, false);
    acc = __builtin_amdgcn_wmma_f32_16x16x32_f16(false, a_hi.v, false, bf_hi.v,
                                                 (short)0, acc, false, false);

    if (w >= 4) {   // publish z/n gate tiles; r tiles stay in waves 0-3 regs
#pragma unroll
      for (int j = 0; j < 8; ++j) lds_gh[rbase + j][g - 64] = acc[j];
    }

    // ---- x staging at 64-step tile boundaries (waves 8..11) --------------
    if ((t & 63) == 0 && w >= 8) {
      int q = tid - 256;
#if HAVE_ASYNC_LDS
      __builtin_amdgcn_s_wait_asynccnt(0);   // current tile resident in LDS
      if (t + 64 < Tt) {                     // prefetch next tile, other buf
        int nbuf = ((t >> 6) + 1) & 1;
#pragma unroll
        for (int i = 0; i < 8; ++i) {
          int e = q * 8 + i, row = e & 15, tt = e >> 4;
          async_g2l_b32(x + (b0 + row) * Tt + (t + 64) + tt,
                        &lds_x[nbuf][tt][row]);
        }
      }
#else
#pragma unroll
      for (int i = 0; i < 8; ++i) {
        int e = q * 8 + i, row = e & 15, tt = e >> 4;
        lds_x[0][tt][row] = x[(b0 + row) * Tt + t + tt];
      }
#endif
    }

    lds_barrier();   // gh tiles + staged x visible; STOREcnt not drained

    // ---- phase B: gate math + state update (waves 0..3) ------------------
    if (w < 4) {
#if HAVE_ASYNC_LDS
      const int xb = (t >> 6) & 1;
#else
      const int xb = 0;
#endif
#pragma unroll
      for (int j = 0; j < 8; ++j) {
        int r = rbase + j;                           // batch row in tile
        float xv   = lds_x[xb][t & 63][r];
        float rg   = sigmoid_(xv * wir + bir  + acc[j]);           // r gate
        float zg   = sigmoid_(xv * wiz + biz  + lds_gh[r][c]);     // z gate
        float ng   = tanh_  (xv * win + bin_ + rg * lds_gh[r][64 + c]);
        float hnew = ng + zg * (hreg[j] - ng);       // (1-z)*n + z*h
        hreg[j] = hnew;
        lds_h16[r][c] = (_Float16)hnew;              // f16 copy for next A-frag
        states[((b0 + r) * Tt + t) * Hh + c] = hnew; // streamed to HBM (async)
      }
    }
    lds_barrier();   // h(t) visible to all waves' A-fragment loads
  }
}

// ---------------------------------------------------------------------------
// Head: out[b,t] = dot(states[b,t,:], Wout) + bout + x[b,t]. Pure bandwidth
// (~130 MB read -> ~6 us at 23.3 TB/s), float4-vectorized row-per-thread.
// ---------------------------------------------------------------------------
__global__ void __launch_bounds__(256) gru_head_kernel(
    const float* __restrict__ states, const float* __restrict__ x,
    const float* __restrict__ Wout,   const float* __restrict__ bout,
    float* __restrict__ out)
{
  int i = blockIdx.x * 256 + threadIdx.x;     // (b,t) row index
  if (i >= Bb * Tt) return;
  const float4* row = (const float4*)(states + (size_t)i * Hh);
  float s = 0.0f;
#pragma unroll
  for (int q = 0; q < 16; ++q) {
    float4 v = row[q];
    s += v.x * Wout[q * 4 + 0] + v.y * Wout[q * 4 + 1] +
         v.z * Wout[q * 4 + 2] + v.w * Wout[q * 4 + 3];
  }
  out[i] = s + bout[0] + x[i];
}

extern "C" void kernel_launch(void* const* d_in, const int* in_sizes, int n_in,
                              void* d_out, int out_size, void* d_ws, size_t ws_size,
                              hipStream_t stream) {
  (void)in_sizes; (void)n_in; (void)out_size; (void)d_ws; (void)ws_size;
  const float* x    = (const float*)d_in[0];
  const float* Wih  = (const float*)d_in[1];
  const float* Whh  = (const float*)d_in[2];
  const float* bih  = (const float*)d_in[3];
  const float* bhh  = (const float*)d_in[4];
  const float* Wout = (const float*)d_in[5];
  const float* bout = (const float*)d_in[6];

  float* out    = (float*)d_out;          // tuple: out [B*T] then states [B*T*H]
  float* states = out + Bb * Tt;

  gru_scan_kernel<<<dim3(2), dim3(TB), 0, stream>>>(x, Wih, Whh, bih, bhh, states);
  gru_head_kernel<<<dim3((Bb * Tt + 255) / 256), dim3(256), 0, stream>>>(
      states, x, Wout, bout, out);
}
